// MultiModalRelationGraph_34041910788303
// MI455X (gfx1250) — compile-verified, matrix operand order checked by reading, not run on hardware
//
#include <hip/hip_runtime.h>

// ---------------------------------------------------------------------------
// MultiModalRelationGraph for MI455X (gfx1250, wave32, WMMA bf16 path)
//
// Dense GEMMs (~247 GFLOP total) run on v_wmma_f32_16x16x32_bf16 with a
// 32x32 register tile per wave (2x2 WMMA tiles -> A/B fragment reuse, 2x
// the arithmetic intensity of the 16x16 version).
// Graph aggregation (86k edges) uses f32 global atomics (memory-bound noise).
// ---------------------------------------------------------------------------

typedef __attribute__((ext_vector_type(16))) __bf16 v16bf;
typedef __attribute__((ext_vector_type(8)))  __bf16 v8bf;
typedef __attribute__((ext_vector_type(8)))  float  v8f;

__device__ __forceinline__ unsigned short f2bf(float f) {
    unsigned u = __float_as_uint(f);
    unsigned r = u + 0x7FFFu + ((u >> 16) & 1u);   // round-to-nearest-even
    return (unsigned short)(r >> 16);
}

// monotone float <-> uint encoding for atomicMax-based segment max
__device__ __forceinline__ unsigned enc_f(float f) {
    unsigned u = __float_as_uint(f);
    return (u >> 31) ? ~u : (u | 0x80000000u);
}
__device__ __forceinline__ float dec_u(unsigned k) {
    unsigned u = (k >> 31) ? (k & 0x7FFFFFFFu) : ~k;
    return __uint_as_float(u);
}
#define ENC_NEG_INF 0x007FFFFFu   // enc(-inf)

__device__ __forceinline__ v16bf frag16(const unsigned short* p) {
    v8bf lo = *(const v8bf*)(p);
    v8bf hi = *(const v8bf*)(p + 16);
    return __builtin_shufflevector(lo, hi, 0,1,2,3,4,5,6,7,8,9,10,11,12,13,14,15);
}

// --------------------------- conversion kernels ----------------------------

__global__ void mmg_cvt_bf16(const float* __restrict__ src,
                             unsigned short* __restrict__ dst, size_t n) {
    size_t i = (size_t)blockIdx.x * blockDim.x + threadIdx.x;
    if (i < n) dst[i] = f2bf(src[i]);
}

// W[K,Nn] row-major fp32  ->  Wt[Nn,K] row-major bf16
__global__ void mmg_tcvt_bf16(const float* __restrict__ W,
                              unsigned short* __restrict__ Wt, int K, int Nn) {
    size_t i = (size_t)blockIdx.x * blockDim.x + threadIdx.x;
    if (i < (size_t)K * Nn) {
        int k = (int)(i % K);
        int n = (int)(i / K);
        Wt[i] = f2bf(W[(size_t)k * Nn + n]);
    }
}

// ------------------------------ WMMA GEMM ----------------------------------
// C[M,Nn] = A[M,K](bf16) @ Wt[Nn,K]^T(bf16) (+bias).
// One wave computes a 32x32 output tile = 2x2 WMMA 16x16 tiles, reusing both
// A fragments and B fragments across the 2x2 grid (4 WMMAs per 8 b128 loads).
// Fragment layout per ISA 16-bit A 16x32: lane<16 -> K {0..7,16..23},
// lane>=16 -> K {8..15,24..31}; B mirrored with per-lane column.

__global__ void __launch_bounds__(256)
mmg_gemm_bf16(const unsigned short* __restrict__ A,
              const unsigned short* __restrict__ Wt,
              const float* __restrict__ bias,
              float* __restrict__ C,
              unsigned short* __restrict__ Cbf,
              int M, int Nn, int K) {
    int wave = blockIdx.x * 8 + (threadIdx.x >> 5);
    int lane = threadIdx.x & 31;
    int tilesN = Nn >> 5;                 // 32-wide tile groups
    int tiles  = (M >> 5) * tilesN;
    if (wave >= tiles) return;
    int tm = wave / tilesN;
    int tn = wave - tm * tilesN;
    int l15  = lane & 15;
    int koff = (lane < 16) ? 0 : 8;

    int row0 = tm * 32 + l15;
    int col0 = tn * 32 + l15;
    const unsigned short* ap0 = A  + (size_t)row0 * K + koff;
    const unsigned short* ap1 = ap0 + (size_t)16 * K;
    const unsigned short* bp0 = Wt + (size_t)col0 * K + koff;
    const unsigned short* bp1 = bp0 + (size_t)16 * K;

    v8f acc00 = {0.f,0.f,0.f,0.f,0.f,0.f,0.f,0.f};
    v8f acc01 = acc00, acc10 = acc00, acc11 = acc00;

    for (int k0 = 0; k0 < K; k0 += 32) {
        v16bf a0 = frag16(ap0 + k0);
        v16bf a1 = frag16(ap1 + k0);
        v16bf b0 = frag16(bp0 + k0);
        v16bf b1 = frag16(bp1 + k0);
        acc00 = __builtin_amdgcn_wmma_f32_16x16x32_bf16(false, a0, false, b0, (short)0, acc00, false, false);
        acc01 = __builtin_amdgcn_wmma_f32_16x16x32_bf16(false, a0, false, b1, (short)0, acc01, false, false);
        acc10 = __builtin_amdgcn_wmma_f32_16x16x32_bf16(false, a1, false, b0, (short)0, acc10, false, false);
        acc11 = __builtin_amdgcn_wmma_f32_16x16x32_bf16(false, a1, false, b1, (short)0, acc11, false, false);
    }

    // C/D layout: VGPR r -> (M = mtile*16 + r (+8 for upper lanes), N = col)
    int rb = (lane < 16) ? 0 : 8;
    int mbase = tm * 32;
    v8f accs[4] = {acc00, acc01, acc10, acc11};
#pragma unroll
    for (int t = 0; t < 4; ++t) {
        int mt = mbase + (t >> 1) * 16 + rb;
        int cn = tn * 32 + ((t & 1) ? 16 : 0) + l15;
        float bv = bias ? bias[cn] : 0.f;
#pragma unroll
        for (int r = 0; r < 8; ++r) {
            float v = accs[t][r] + bv;
            size_t o = (size_t)(mt + r) * Nn + cn;
            if (C)   C[o]   = v;
            if (Cbf) Cbf[o] = f2bf(v);
        }
    }
}

// ---------------------- attention coefficient kernel -----------------------
// a_s[n,h] = sum_c h[n,h*256+c]*Asrc[h,c] ; a_d likewise. One wave per (n,h).

__global__ void mmg_attn(const float* __restrict__ h,
                         const float* __restrict__ Asrc,
                         const float* __restrict__ Adst,
                         float* __restrict__ a_s, float* __restrict__ a_d, int N) {
    int gw   = (int)(((size_t)blockIdx.x * blockDim.x + threadIdx.x) >> 5);
    int lane = threadIdx.x & 31;
    if (gw >= N * 4) return;
    int n = gw >> 2, hd = gw & 3;
    const float* hp = h + (size_t)n * 1024 + hd * 256;
    const float* ws = Asrc + hd * 256;
    const float* wd = Adst + hd * 256;
    float ss = 0.f, sd = 0.f;
    for (int i = lane; i < 256; i += 32) {
        float v = hp[i];
        ss += v * ws[i];
        sd += v * wd[i];
    }
    for (int o = 16; o; o >>= 1) {
        ss += __shfl_down(ss, o, 32);
        sd += __shfl_down(sd, o, 32);
    }
    if (lane == 0) { a_s[n * 4 + hd] = ss; a_d[n * 4 + hd] = sd; }
}

// ------------------------- per-layer init / passes -------------------------

__global__ void mmg_init_layer(float* __restrict__ numer, float* __restrict__ denom,
                               unsigned* __restrict__ mkey, int N) {
    size_t i = (size_t)blockIdx.x * blockDim.x + threadIdx.x;
    if (i < (size_t)N * 1024) numer[i] = 0.f;
    if (i < (size_t)N * 4) { denom[i] = 0.f; mkey[i] = ENC_NEG_INF; }
}

__global__ void mmg_edge_max(const int* __restrict__ ei, int E, int N,
                             const float* __restrict__ a_s, const float* __restrict__ a_d,
                             unsigned* __restrict__ mkey) {
    size_t i = (size_t)blockIdx.x * blockDim.x + threadIdx.x;
    if (i >= (size_t)(E + N) * 4) return;
    int edge = (int)(i >> 2), hd = (int)(i & 3);
    int src, dst;
    if (edge < E) { src = ei[edge]; dst = ei[E + edge]; }
    else          { src = dst = edge - E; }
    float e = a_s[src * 4 + hd] + a_d[dst * 4 + hd];
    e = (e > 0.f) ? e : 0.2f * e;                    // leaky_relu(0.2)
    atomicMax(&mkey[dst * 4 + hd], enc_f(e));
}

// one block (256 threads) per edge; thread c handles channel c for all heads
__global__ void mmg_edge_scatter(const int* __restrict__ ei, int E, int N,
                                 const float* __restrict__ a_s, const float* __restrict__ a_d,
                                 const unsigned* __restrict__ mkey,
                                 float* __restrict__ denom,
                                 const float* __restrict__ h,
                                 float* __restrict__ numer) {
    int edge = blockIdx.x;
    int c = threadIdx.x;
    int src, dst;
    if (edge < E) { src = ei[edge]; dst = ei[E + edge]; }
    else          { src = dst = edge - E; }
#pragma unroll
    for (int hd = 0; hd < 4; ++hd) {
        float e = a_s[src * 4 + hd] + a_d[dst * 4 + hd];
        e = (e > 0.f) ? e : 0.2f * e;
        float m = dec_u(mkey[dst * 4 + hd]);
        float w = __expf(e - m);
        float val = w * h[(size_t)src * 1024 + hd * 256 + c];
        atomicAdd(&numer[(size_t)dst * 1024 + hd * 256 + c], val);
        if (c == hd) atomicAdd(&denom[dst * 4 + hd], w);
    }
}

__global__ void mmg_fin_concat(const float* __restrict__ numer, const float* __restrict__ denom,
                               const float* __restrict__ bias,
                               unsigned short* __restrict__ xbf, int N) {
    size_t i = (size_t)blockIdx.x * blockDim.x + threadIdx.x;
    if (i >= (size_t)N * 1024) return;
    int n = (int)(i >> 10), j = (int)(i & 1023), hd = j >> 8;
    float v = numer[i] / (denom[n * 4 + hd] + 1e-16f) + bias[j];
    xbf[i] = f2bf(v);
}

__global__ void mmg_fin_mean(const float* __restrict__ numer, const float* __restrict__ denom,
                             const float* __restrict__ bias,
                             float* __restrict__ fin, int N) {
    size_t i = (size_t)blockIdx.x * blockDim.x + threadIdx.x;
    if (i >= (size_t)N * 256) return;
    int n = (int)(i >> 8), c = (int)(i & 255);
    float s = 0.f;
#pragma unroll
    for (int hd = 0; hd < 4; ++hd)
        s += numer[(size_t)n * 1024 + hd * 256 + c] / (denom[n * 4 + hd] + 1e-16f);
    fin[i] = 0.25f * s + bias[c];
}

// --------------------------- LayerNorm + mean ------------------------------

__global__ void mmg_zero_out(float* __restrict__ out) { out[threadIdx.x] = 0.f; }

__global__ void mmg_ln_mean(const float* __restrict__ x, const float* __restrict__ g,
                            const float* __restrict__ b, float* __restrict__ out, int N) {
    __shared__ float red[256];
    int c = threadIdx.x;
    float accum = 0.f;
    for (int n = blockIdx.x; n < N; n += gridDim.x) {
        float v = x[(size_t)n * 256 + c];
        red[c] = v; __syncthreads();
        for (int s = 128; s > 0; s >>= 1) { if (c < s) red[c] += red[c + s]; __syncthreads(); }
        float mu = red[0] * (1.f / 256.f); __syncthreads();
        float d = v - mu;
        red[c] = d * d; __syncthreads();
        for (int s = 128; s > 0; s >>= 1) { if (c < s) red[c] += red[c + s]; __syncthreads(); }
        float var = red[0] * (1.f / 256.f); __syncthreads();
        accum += (d * rsqrtf(var + 1e-5f)) * g[c] + b[c];
    }
    atomicAdd(&out[c], accum / (float)N);
}

// ------------------------------- launcher ----------------------------------

extern "C" void kernel_launch(void* const* d_in, const int* in_sizes, int n_in,
                              void* d_out, int out_size, void* d_ws, size_t ws_size,
                              hipStream_t stream) {
    (void)n_in; (void)out_size; (void)ws_size;
    const int NR = 4, BATCH = 4, T_R = 2048, T_A = 4096;
    const int D_REG = 512, D_AUD = 768, HID = 256;
    const int ROWS_R = BATCH * T_R;          // 8192
    const int ROWS_A = BATCH * T_A;          // 16384
    const int NN = NR * ROWS_R + ROWS_A;     // 49152
    const int DH = 4 * HID;                  // 1024

    // ---- inputs (JAX pytree / sorted-key flattening order) ----
    const float* reg_in[4] = { (const float*)d_in[0], (const float*)d_in[1],
                               (const float*)d_in[2], (const float*)d_in[3] };  // mouth,le,re,nose
    const float* audio    = (const float*)d_in[4];
    const float* audio_W  = (const float*)d_in[5];
    const float* audio_b  = (const float*)d_in[6];
    const float* gat_W[3]  = { (const float*)d_in[7],  (const float*)d_in[11], (const float*)d_in[15] };
    const float* gat_ad[3] = { (const float*)d_in[8],  (const float*)d_in[12], (const float*)d_in[16] };
    const float* gat_as[3] = { (const float*)d_in[9],  (const float*)d_in[13], (const float*)d_in[17] };
    const float* gat_b[3]  = { (const float*)d_in[10], (const float*)d_in[14], (const float*)d_in[18] };
    const float* ln_b = (const float*)d_in[19];
    const float* ln_g = (const float*)d_in[20];
    // proj params flattened in sorted name order: left_eye, mouth, nose, right_eye
    // -> remap to REGIONS order mouth, left_eye, right_eye, nose
    const float* proj_W[4] = { (const float*)d_in[23], (const float*)d_in[21],
                               (const float*)d_in[27], (const float*)d_in[25] };
    const float* proj_b[4] = { (const float*)d_in[24], (const float*)d_in[22],
                               (const float*)d_in[28], (const float*)d_in[26] };
    const int* ei = (const int*)d_in[29];
    const int E = in_sizes[29] / 2;

    // ---- workspace layout ----
    char* ws = (char*)d_ws;
    size_t off = 0;
    auto take = [&](size_t bytes) -> char* {
        char* p = ws + off;
        off = (off + bytes + 255) & ~(size_t)255;
        return p;
    };
    unsigned short* act_reg[4];
    for (int r = 0; r < 4; ++r) act_reg[r] = (unsigned short*)take((size_t)ROWS_R * D_REG * 2);
    unsigned short* act_aud = (unsigned short*)take((size_t)ROWS_A * D_AUD * 2);
    unsigned short* wt_proj[4];
    for (int r = 0; r < 4; ++r) wt_proj[r] = (unsigned short*)take((size_t)HID * D_REG * 2);
    unsigned short* wt_aud = (unsigned short*)take((size_t)HID * D_AUD * 2);
    unsigned short* wt_gat[3];
    wt_gat[0] = (unsigned short*)take((size_t)DH * HID * 2);
    wt_gat[1] = (unsigned short*)take((size_t)DH * DH * 2);
    wt_gat[2] = (unsigned short*)take((size_t)DH * DH * 2);
    unsigned short* xbf = (unsigned short*)take((size_t)NN * DH * 2);
    float* hbuf  = (float*)take((size_t)NN * DH * 4);
    float* numer = (float*)take((size_t)NN * DH * 4);
    float* a_s   = (float*)take((size_t)NN * 4 * 4);
    float* a_d   = (float*)take((size_t)NN * 4 * 4);
    unsigned* mkey = (unsigned*)take((size_t)NN * 4 * 4);
    float* denom = (float*)take((size_t)NN * 4 * 4);
    float* fin   = (float*)take((size_t)NN * HID * 4);

    auto nb = [](size_t n) { return (unsigned)((n + 255) / 256); };

    // ---- 1) convert activations & weights to bf16 (weights transposed) ----
    for (int r = 0; r < 4; ++r)
        mmg_cvt_bf16<<<nb((size_t)ROWS_R * D_REG), 256, 0, stream>>>(reg_in[r], act_reg[r], (size_t)ROWS_R * D_REG);
    mmg_cvt_bf16<<<nb((size_t)ROWS_A * D_AUD), 256, 0, stream>>>(audio, act_aud, (size_t)ROWS_A * D_AUD);
    for (int r = 0; r < 4; ++r)
        mmg_tcvt_bf16<<<nb((size_t)D_REG * HID), 256, 0, stream>>>(proj_W[r], wt_proj[r], D_REG, HID);
    mmg_tcvt_bf16<<<nb((size_t)D_AUD * HID), 256, 0, stream>>>(audio_W, wt_aud, D_AUD, HID);
    mmg_tcvt_bf16<<<nb((size_t)HID * DH), 256, 0, stream>>>(gat_W[0], wt_gat[0], HID, DH);
    mmg_tcvt_bf16<<<nb((size_t)DH * DH), 256, 0, stream>>>(gat_W[1], wt_gat[1], DH, DH);
    mmg_tcvt_bf16<<<nb((size_t)DH * DH), 256, 0, stream>>>(gat_W[2], wt_gat[2], DH, DH);

    auto gemm = [&](const unsigned short* A, const unsigned short* Wt, const float* bias,
                    float* C, unsigned short* Cbf, int M, int Nn, int K) {
        int tiles = (M / 32) * (Nn / 32);
        mmg_gemm_bf16<<<nb((size_t)tiles * 32), 256, 0, stream>>>(A, Wt, bias, C, Cbf, M, Nn, K);
    };

    // ---- 2) projections -> x bf16 [NN, 256] ----
    for (int r = 0; r < 4; ++r)
        gemm(act_reg[r], wt_proj[r], proj_b[r], nullptr,
             xbf + (size_t)r * ROWS_R * HID, ROWS_R, HID, D_REG);
    gemm(act_aud, wt_aud, audio_b, nullptr,
         xbf + (size_t)NR * ROWS_R * HID, ROWS_A, HID, D_AUD);

    // ---- 3) GAT layers ----
    int din = HID;
    for (int l = 0; l < 3; ++l) {
        gemm(xbf, wt_gat[l], nullptr, hbuf, nullptr, NN, DH, din);
        mmg_attn<<<nb((size_t)NN * 4 * 32), 256, 0, stream>>>(hbuf, gat_as[l], gat_ad[l], a_s, a_d, NN);
        mmg_init_layer<<<nb((size_t)NN * DH), 256, 0, stream>>>(numer, denom, mkey, NN);
        mmg_edge_max<<<nb((size_t)(E + NN) * 4), 256, 0, stream>>>(ei, E, NN, a_s, a_d, mkey);
        mmg_edge_scatter<<<(unsigned)(E + NN), 256, 0, stream>>>(ei, E, NN, a_s, a_d, mkey, denom, hbuf, numer);
        if (l < 2) {
            mmg_fin_concat<<<nb((size_t)NN * DH), 256, 0, stream>>>(numer, denom, gat_b[l], xbf, NN);
            din = DH;
        } else {
            mmg_fin_mean<<<nb((size_t)NN * HID), 256, 0, stream>>>(numer, denom, gat_b[2], fin, NN);
        }
    }

    // ---- 4) LayerNorm + mean over nodes -> out[1,256] ----
    mmg_zero_out<<<1, 256, 0, stream>>>((float*)d_out);
    mmg_ln_mean<<<256, 256, 0, stream>>>(fin, ln_g, ln_b, (float*)d_out, NN);
}
